// Net_35734127902991
// MI455X (gfx1250) — matrix operand order repaired and checked by reading, not compiled, not run on hardware
//
#include <hip/hip_runtime.h>
#include <hip/hip_bf16.h>

// ---------------------------------------------------------------------------
// Label propagation (dense normalized adjacency + cosine-sim fallback)
// N=4096 nodes, D=256 features, C=16 classes, E=65536 edges.
// GEMMs (sim = xn@xn^T and z = adj@tl) run on v_wmma_f32_16x16x32_bf16.
// C=16 == one WMMA N tile; propagation K-splits 8 waves/strip + LDS reduce.
// ---------------------------------------------------------------------------

#define NN   4096
#define DD   256
#define CC   16
#define EE   65536
#define ITERS 64

typedef __bf16 bf16_t;
typedef __attribute__((ext_vector_type(16))) __bf16 v16bf;
typedef __attribute__((ext_vector_type(8)))  float  v8f;

// ctrl layout (ints unless noted):
// [0]=active [1]=bad [2]=thr(float bits) [3]=pre_nwl [4]=pre_fea
// [5]=nwl [6]=power(1+bad) [7]=fea_flag [8]=fea_stag

// ------------------------------- setup -------------------------------------

__global__ void k_scatter(const int* __restrict__ ei, bf16_t* __restrict__ adj) {
  int e = blockIdx.x * blockDim.x + threadIdx.x;
  if (e >= EE) return;
  int r = ei[e], c = ei[EE + e];
  adj[(size_t)r * NN + c] = (bf16_t)1.0f;
  adj[(size_t)c * NN + r] = (bf16_t)1.0f;   // undirected: max(A, A^T)
}

__global__ void k_deg(const bf16_t* __restrict__ adj, float* __restrict__ dinv) {
  __shared__ float red[256];
  int row = blockIdx.x, t = threadIdx.x;
  float s = 0.f;
  for (int j = t; j < NN; j += 256) s += (float)adj[(size_t)row * NN + j];
  red[t] = s; __syncthreads();
  for (int o = 128; o > 0; o >>= 1) { if (t < o) red[t] += red[t + o]; __syncthreads(); }
  if (t == 0) dinv[row] = (red[0] > 0.f) ? rsqrtf(red[0]) : 0.f;
}

__global__ void k_norm_adj(bf16_t* __restrict__ adj, const float* __restrict__ dinv) {
  size_t idx = (size_t)blockIdx.x * blockDim.x + threadIdx.x;
  if (idx >= (size_t)NN * NN) return;
  int i = (int)(idx >> 12), j = (int)(idx & (NN - 1));
  if ((float)adj[idx] != 0.f) adj[idx] = (bf16_t)(dinv[i] * dinv[j]);
}

__global__ void k_xnorm(const float* __restrict__ x, bf16_t* __restrict__ xn) {
  __shared__ float red[256];
  int row = blockIdx.x, t = threadIdx.x;          // blockDim == DD == 256
  float v = x[row * DD + t];
  red[t] = v * v; __syncthreads();
  for (int o = 128; o > 0; o >>= 1) { if (t < o) red[t] += red[t + o]; __syncthreads(); }
  float rn = rsqrtf(red[0]);
  xn[row * DD + t] = (bf16_t)(v * rn);
}

// sim = xn @ xn^T (bf16 in, f32 acc, bf16 out), diagonal zeroed.
// 128-thread block = 4 waves, each wave one 16x16 tile; K-loop over D=256.
__global__ void k_sim_wmma(const bf16_t* __restrict__ xn, bf16_t* __restrict__ sim) {
  int wave = threadIdx.x >> 5, lane = threadIdx.x & 31;
  int m0 = blockIdx.x * 16;
  int n0 = (blockIdx.y * 4 + wave) * 16;
  int rowA = m0 + (lane & 15);
  int khA  = (lane >> 4) * 16;                     // A: lanes 0-15 K 0-15; 16-31 K 16-31
  int colB = n0 + (lane & 15);
  int kbB  = (lane >> 4) * 16;                     // B: lanes 0-15 K 0-15; 16-31 K 16-31
  v8f acc = {};
  const bf16_t* ap = xn + (size_t)rowA * DD;
  const bf16_t* bp = xn + (size_t)colB * DD;
  for (int k0 = 0; k0 < DD; k0 += 32) {
    __builtin_prefetch(ap + k0 + 32, 0, 0);
    v16bf a, b;
#pragma unroll
    for (int i = 0; i < 16; ++i) {
      int ka = khA + i;
      a[i] = ap[k0 + ka];
      b[i] = bp[k0 + kbB + i];
    }
    acc = __builtin_amdgcn_wmma_f32_16x16x32_bf16(false, a, false, b,
                                                  (short)0, acc, false, false);
  }
  int n  = n0 + (lane & 15);
  int mb = m0 + (lane >> 4) * 8;                   // C/D: vgpr r -> M = r + 8*(lane>=16)
#pragma unroll
  for (int r = 0; r < 8; ++r) {
    int m = mb + r;
    float v = acc[r];
    if (m == n) v = 0.0f;                          // fill_diagonal_(0)
    sim[(size_t)m * NN + n] = (bf16_t)v;
  }
}

__global__ void k_init_tl(const int* __restrict__ y, const int* __restrict__ mask,
                          float* __restrict__ tl) {
  int row = blockIdx.x * blockDim.x + threadIdx.x;
  if (row >= NN) return;
  float m = mask[row] ? 1.f : 0.f;
  int cy = y[row];
  for (int c = 0; c < CC; ++c) tl[row * CC + c] = (c == cy) ? m : 0.f;
}

__global__ void k_ctrl_init(const int* __restrict__ mask, int* __restrict__ ctrl) {
  __shared__ int red[256];
  int t = threadIdx.x, s = 0;
  for (int i = t; i < NN; i += 256) s += (mask[i] ? 1 : 0);
  red[t] = s; __syncthreads();
  for (int o = 128; o > 0; o >>= 1) { if (t < o) red[t] += red[t + o]; __syncthreads(); }
  if (t == 0) {
    ctrl[0] = 1;                       // active
    ctrl[1] = 0;                       // bad
    ((float*)ctrl)[2] = 0.5f;          // thr
    ctrl[3] = NN - red[0];             // pre_nwl
    ctrl[4] = -1;                      // pre_fea sentinel
  }
}

// ----------------------------- loop body -----------------------------------

__global__ void k_ctrl_begin(const float* __restrict__ tl, int* __restrict__ ctrl) {
  __shared__ int red[256];
  int t = threadIdx.x, cnt = 0;
  for (int row = t; row < NN; row += 256) {
    float mx = 0.f;
    for (int c = 0; c < CC; ++c) mx = fmaxf(mx, tl[row * CC + c]);
    if (mx != 0.f) cnt++;
  }
  red[t] = cnt; __syncthreads();
  for (int o = 128; o > 0; o >>= 1) { if (t < o) red[t] += red[t + o]; __syncthreads(); }
  if (t == 0) {
    int nwl = NN - red[0];
    if (ctrl[0] && nwl == 0) ctrl[0] = 0;          // converged -> all later kernels no-op
    if (ctrl[0]) {
      int stag = (ctrl[3] == nwl);
      int bad  = stag ? ctrl[1] + 1 : 0;
      if (!stag) ((float*)ctrl)[2] = 0.5f;
      ctrl[1] = bad;
      ctrl[5] = nwl;
      ctrl[6] = 1 + bad;                           // adj power, 1..3
      ctrl[7] = (bad > 1) ? 1 : 0;                 // fea branch flag
      ctrl[8] = (ctrl[4] == nwl) ? 1 : 0;          // fea_stag
      ctrl[3] = nwl;                               // pre_nwl = nwl
    }
  }
}

// f32 (N x C) -> bf16 transposed (C x N) so WMMA B loads are contiguous in K.
__global__ void k_conv_bf16T(const float* __restrict__ src, bf16_t* __restrict__ dstT) {
  int i = blockIdx.x * blockDim.x + threadIdx.x;   // NN*CC elements
  int k = i >> 4, n = i & 15;
  dstT[n * NN + k] = (bf16_t)src[i];
}

// z_out(16 rows x 16 cls) = adj(16 x 4096) @ tl(4096 x 16).
// 256 threads = 8 waves per strip; K split 8 ways, LDS partial-tile reduce.
__global__ void k_prop_wmma(const bf16_t* __restrict__ adj, const bf16_t* __restrict__ tlT,
                            const float* __restrict__ zin, float* __restrict__ zout,
                            const int* __restrict__ ctrl, int pass) {
  __shared__ float part[8][16 * CC];
  int tid = threadIdx.x;
  int wave = tid >> 5, lane = tid & 31;
  int m0 = blockIdx.x * 16;
  if (!ctrl[0] || pass >= ctrl[6]) {               // inactive / power reached: pass-through
    for (int i = tid; i < 16 * CC; i += 256) zout[m0 * CC + i] = zin[m0 * CC + i];
    return;
  }
  int rowA = m0 + (lane & 15);
  int khA  = (lane >> 4) * 16;
  int nB   = lane & 15;
  int kbB  = (lane >> 4) * 16;
  v8f acc = {};
  const bf16_t* ap = adj + (size_t)rowA * NN;
  const bf16_t* bp = tlT + (size_t)nB * NN;
  int kBeg = wave * (NN / 8), kEnd = kBeg + NN / 8;
  for (int k0 = kBeg; k0 < kEnd; k0 += 32) {
    __builtin_prefetch(ap + k0 + 64, 0, 0);
    v16bf a, b;
#pragma unroll
    for (int i = 0; i < 16; ++i) {
      int ka = khA + i;
      a[i] = ap[k0 + ka];
      b[i] = bp[k0 + kbB + i];
    }
    acc = __builtin_amdgcn_wmma_f32_16x16x32_bf16(false, a, false, b,
                                                  (short)0, acc, false, false);
  }
  int mloc = (lane >> 4) * 8;                      // local M base for this lane
#pragma unroll
  for (int r = 0; r < 8; ++r) part[wave][(mloc + r) * CC + (lane & 15)] = acc[r];
  __syncthreads();
  float s = 0.f;                                   // one output element per thread
#pragma unroll
  for (int w = 0; w < 8; ++w) s += part[w][tid];
  zout[m0 * CC + tid] = s;
}

__global__ void k_update(const float* __restrict__ z, float* __restrict__ tl,
                         const int* __restrict__ ctrl) {
  if (!ctrl[0]) return;
  int row = blockIdx.x * blockDim.x + threadIdx.x;
  if (row >= NN) return;
  float mx = 0.f, zmx = -1e30f;
  for (int c = 0; c < CC; ++c) {
    mx  = fmaxf(mx,  tl[row * CC + c]);
    zmx = fmaxf(zmx, z [row * CC + c]);
  }
  bool known = (mx != 0.f);
  if (zmx >= 0.5f && !known)
    for (int c = 0; c < CC; ++c) tl[row * CC + c] = (z[row * CC + c] >= 0.5f) ? 1.f : 0.f;
}

__global__ void k_copy_f32(const float* __restrict__ s, float* __restrict__ d, int n) {
  int i = blockIdx.x * blockDim.x + threadIdx.x;
  if (i < n) d[i] = s[i];
}

// feature-similarity fallback: tl[i] = tl_in[argmax_j sim'(i,j)] when smax >= thr2
__global__ void k_fea(const bf16_t* __restrict__ sim, const float* __restrict__ tl_in,
                      float* __restrict__ tl, const int* __restrict__ ctrl) {
  if (!ctrl[0] || !ctrl[7]) return;
  __shared__ float smax[128]; __shared__ int sidx[128];
  int row = blockIdx.x, t = threadIdx.x;
  int fea_stag = ctrl[8];
  float thr = ((const float*)ctrl)[2];
  float bmax = -1e30f; int bidx = 0;
  for (int j = t; j < NN; j += 128) {
    float s = (float)sim[(size_t)row * NN + j];
    if (fea_stag) s = (s < thr) ? s : (thr - 0.1f);
    if (s > bmax) { bmax = s; bidx = j; }
  }
  smax[t] = bmax; sidx[t] = bidx; __syncthreads();
  for (int o = 64; o > 0; o >>= 1) {
    if (t < o) {
      if (smax[t + o] > smax[t] ||
          (smax[t + o] == smax[t] && sidx[t + o] < sidx[t])) {
        smax[t] = smax[t + o]; sidx[t] = sidx[t + o];
      }
    }
    __syncthreads();
  }
  float thr2 = fea_stag ? (thr - 0.1f) : 0.5f;
  float mx = 0.f;
  for (int c = 0; c < CC; ++c) mx = fmaxf(mx, tl_in[row * CC + c]);
  bool known = (mx != 0.f);
  if (t < CC && smax[0] >= thr2 && !known)
    tl[row * CC + t] = tl_in[sidx[0] * CC + t];
}

__global__ void k_ctrl_end(int* __restrict__ ctrl) {
  if (blockIdx.x == 0 && threadIdx.x == 0) {
    if (ctrl[0] && ctrl[7]) {
      float thr = ((float*)ctrl)[2];
      ((float*)ctrl)[2] = ctrl[8] ? (thr - 0.1f) : 0.5f;
      ctrl[4] = ctrl[5];                           // pre_fea = nwl
      ctrl[1] = 0;                                 // bad reset after fea branch
    }
  }
}

__global__ void k_out(const float* __restrict__ tl, const float* __restrict__ x,
                      float* __restrict__ out) {
  size_t i = (size_t)blockIdx.x * blockDim.x + threadIdx.x;
  const size_t n1 = (size_t)NN * CC, n2 = (size_t)NN * DD;
  if (i < n1) out[i] = tl[i];
  else if (i < n1 + n2) out[i] = x[i - n1];
}

// ------------------------------- launcher ----------------------------------

extern "C" void kernel_launch(void* const* d_in, const int* in_sizes, int n_in,
                              void* d_out, int out_size, void* d_ws, size_t ws_size,
                              hipStream_t stream) {
  const float* x    = (const float*)d_in[0];
  const int*   ei   = (const int*)d_in[2];
  const int*   mask = (const int*)d_in[3];
  const int*   y    = (const int*)d_in[4];
  float* out = (float*)d_out;

  char* p = (char*)d_ws;
  bf16_t* adj    = (bf16_t*)p; p += (size_t)NN * NN * 2;   // 32 MiB
  bf16_t* sim    = (bf16_t*)p; p += (size_t)NN * NN * 2;   // 32 MiB
  bf16_t* xn     = (bf16_t*)p; p += (size_t)NN * DD * 2;   // 2 MiB
  float*  tl     = (float*)p;  p += (size_t)NN * CC * 4;
  float*  tl_tmp = (float*)p;  p += (size_t)NN * CC * 4;
  float*  zA     = (float*)p;  p += (size_t)NN * CC * 4;
  float*  zB     = (float*)p;  p += (size_t)NN * CC * 4;
  bf16_t* tlT    = (bf16_t*)p; p += (size_t)NN * CC * 2;   // transposed bf16 labels
  float*  dinv   = (float*)p;  p += (size_t)NN * 4;
  int*    ctrl   = (int*)p;    p += 256;

  // --- build normalized adjacency ---
  hipMemsetAsync(adj, 0, (size_t)NN * NN * 2, stream);
  k_scatter<<<EE / 256, 256, 0, stream>>>(ei, adj);
  k_deg<<<NN, 256, 0, stream>>>(adj, dinv);
  k_norm_adj<<<(int)(((size_t)NN * NN) / 256), 256, 0, stream>>>(adj, dinv);

  // --- cosine-similarity matrix via WMMA ---
  k_xnorm<<<NN, DD, 0, stream>>>(x, xn);
  k_sim_wmma<<<dim3(NN / 16, NN / 64), 128, 0, stream>>>(xn, sim);

  // --- labels + control state ---
  k_init_tl<<<NN / 256, 256, 0, stream>>>(y, mask, tl);
  k_ctrl_init<<<1, 256, 0, stream>>>(mask, ctrl);

  // --- fixed-trip propagation loop (kernels self-predicate on ctrl) ---
  for (int it = 0; it < ITERS; ++it) {
    k_ctrl_begin<<<1, 256, 0, stream>>>(tl, ctrl);

    // z = adj^(1+bad) @ tl ; three passes, extra passes degrade to copies
    const float* src[3] = { tl, zA, zB };
    float*       dst[3] = { zA, zB, zA };
    for (int pass = 0; pass < 3; ++pass) {
      k_conv_bf16T<<<NN * CC / 256, 256, 0, stream>>>(src[pass], tlT);
      k_prop_wmma<<<NN / 16, 256, 0, stream>>>(adj, tlT, src[pass], dst[pass], ctrl, pass);
    }
    k_update<<<NN / 256, 256, 0, stream>>>(zA, tl, ctrl);

    // feature-similarity fallback (reads a snapshot of tl)
    k_copy_f32<<<NN * CC / 256, 256, 0, stream>>>(tl, tl_tmp, NN * CC);
    k_fea<<<NN, 128, 0, stream>>>(sim, tl_tmp, tl, ctrl);
    k_ctrl_end<<<1, 1, 0, stream>>>(ctrl);
  }

  // --- outputs: (train_label, x) concatenated ---
  size_t tot = (size_t)NN * CC + (size_t)NN * DD;
  k_out<<<(int)((tot + 255) / 256), 256, 0, stream>>>(tl, x, out);
}
